// Decoder_38680475467871
// MI455X (gfx1250) — matrix operand order repaired and checked by reading, not compiled, not run on hardware
//
#include <hip/hip_runtime.h>

typedef unsigned short u16;
typedef __attribute__((ext_vector_type(16))) __bf16 bf16x16;
typedef __attribute__((ext_vector_type(8)))  float  f32x8;
typedef __attribute__((ext_vector_type(4)))  unsigned int u32x4;
typedef __attribute__((ext_vector_type(8)))  int  i32x8;
typedef __attribute__((ext_vector_type(4)))  int  i32x4;

#define D_MODEL 1024
#define NHEAD   16
#define HEAD    64
#define D_FF    4096
#define SEQ     2048
#define BATCH   2
#define ROWS    (BATCH * SEQ)   // 4096

#if defined(__has_builtin)
#if __has_builtin(__builtin_amdgcn_tensor_load_to_lds) && __has_builtin(__builtin_amdgcn_s_wait_tensorcnt)
#define HAVE_TDM 1
#endif
#endif
#ifndef HAVE_TDM
#define HAVE_TDM 0
#endif

union Frag {
    uint4   q[2];
    bf16x16 v;
};

__device__ __forceinline__ u16 f2bf(float f) {
    unsigned u = __float_as_uint(f);
    unsigned r = u + 0x7FFFu + ((u >> 16) & 1u);
    return (u16)(r >> 16);
}
__device__ __forceinline__ float bf2f(u16 h) {
    return __uint_as_float(((unsigned)h) << 16);
}
__device__ __forceinline__ float gelu_tanh(float x) {
    const float c = 0.7978845608028654f; // sqrt(2/pi)
    float t = tanhf(c * (x + 0.044715f * x * x * x));
    return 0.5f * x * (1.0f + t);
}

// Low 32 bits of a flat LDS pointer == byte offset within the LDS allocation
// (the shared aperture occupies addr[63:32]; FLAT->LDS keeps addr[31:0]).
__device__ __forceinline__ unsigned lds_off(const void* p) {
    return (unsigned)(uintptr_t)p;
}

#if HAVE_TDM
// Issue one TDM 2-D tile load: global (row-major, bf16, row stride in elems)
// -> LDS. Descriptor packing per CDNA5 ISA §8.3 (group 0) / §8.4 (group 1).
// tensor dims == tile dims (tiles never go OOB here); stride may exceed dim0.
__device__ __forceinline__ void tdm_load_2d(unsigned ldsOff, const u16* gptr,
                                            unsigned d0, unsigned d1,
                                            unsigned long long strideElems) {
    unsigned long long ga = (unsigned long long)(uintptr_t)gptr;
    u32x4 g0;
    g0[0] = 1u;                                   // count=1, user mode
    g0[1] = ldsOff;                               // lds_addr
    g0[2] = (unsigned)ga;                         // global_addr[31:0]
    g0[3] = (unsigned)(ga >> 32) | (2u << 30);    // global_addr[56:32] | type=2
    i32x8 g1;
    g1[0] = (int)(1u << 16);                                  // data_size=1 -> 2B
    g1[1] = (int)((d0 & 0xFFFFu) << 16);                      // tensor_dim0 lo
    g1[2] = (int)((d0 >> 16) | ((d1 & 0xFFFFu) << 16));       // dim0 hi | dim1 lo
    g1[3] = (int)((d1 >> 16) | ((d0 & 0xFFFFu) << 16));       // dim1 hi | tile_dim0
    g1[4] = (int)(d1 & 0xFFFFu);                              // tile_dim1 | tile_dim2=0
    g1[5] = (int)(unsigned)(strideElems & 0xFFFFFFFFull);     // dim0_stride lo
    g1[6] = (int)(unsigned)((strideElems >> 32) & 0xFFFFull); // dim0_stride hi | dim1_stride=0
    g1[7] = 0;
    i32x4 z4 = {0, 0, 0, 0};
#if __clang_major__ >= 23
    i32x8 z8 = {0, 0, 0, 0, 0, 0, 0, 0};
    __builtin_amdgcn_tensor_load_to_lds(g0, g1, z4, z4, z8, 0);
#else
    __builtin_amdgcn_tensor_load_to_lds(g0, g1, z4, z4, 0);
#endif
}
#endif

// ---------------------------------------------------------------------------
// Weight preparation: f32 -> bf16, [N][K] (K contiguous) layout for WMMA B.
// ---------------------------------------------------------------------------
__global__ __launch_bounds__(256) void prep_qkv_w(const float* __restrict__ wq,
                                                  const float* __restrict__ wk,
                                                  const float* __restrict__ wv,
                                                  u16* __restrict__ wt) {
    int i = blockIdx.x * 256 + threadIdx.x;      // over 3072*1024
    int n = i >> 10;
    int k = i & 1023;
    int sel = n >> 10;
    int hn  = n & 1023;
    int h = hn >> 6, e = hn & 63;
    const float* src = (sel == 0) ? wq : (sel == 1) ? wk : wv;
    wt[i] = f2bf(src[((size_t)h * D_MODEL + k) * HEAD + e]);
}

__global__ __launch_bounds__(256) void prep_wo(const float* __restrict__ wo,
                                               u16* __restrict__ wot) {
    int i = blockIdx.x * 256 + threadIdx.x;      // over 1024*1024
    int n = i >> 10;
    int k = i & 1023;
    wot[i] = f2bf(wo[(size_t)k * D_MODEL + n]);
}

__global__ __launch_bounds__(256) void convert_f32_bf16(const float* __restrict__ src,
                                                        u16* __restrict__ dst, int count) {
    int i = blockIdx.x * 256 + threadIdx.x;
    if (i < count) dst[i] = f2bf(src[i]);
}

// ---------------------------------------------------------------------------
// LayerNorm over D=1024, one row per block of 256 threads, bf16 output.
// ---------------------------------------------------------------------------
__global__ __launch_bounds__(256) void ln_kernel(const float* __restrict__ x,
                                                 const float* __restrict__ w,
                                                 const float* __restrict__ b,
                                                 u16* __restrict__ y) {
    __shared__ float r1[256];
    __shared__ float r2[256];
    int row = blockIdx.x;
    int tid = threadIdx.x;
    const float* xr = x + (size_t)row * D_MODEL;
    float4 xv = ((const float4*)xr)[tid];
    float s  = xv.x + xv.y + xv.z + xv.w;
    float ss = xv.x * xv.x + xv.y * xv.y + xv.z * xv.z + xv.w * xv.w;
    r1[tid] = s; r2[tid] = ss;
    __syncthreads();
    for (int st = 128; st > 0; st >>= 1) {
        if (tid < st) { r1[tid] += r1[tid + st]; r2[tid] += r2[tid + st]; }
        __syncthreads();
    }
    float mean = r1[0] * (1.0f / D_MODEL);
    float var  = r2[0] * (1.0f / D_MODEL) - mean * mean;
    float rstd = rsqrtf(var + 1e-5f);
    float vals[4] = {xv.x, xv.y, xv.z, xv.w};
    u16* yr = y + (size_t)row * D_MODEL;
    for (int i = 0; i < 4; ++i) {
        int c = tid * 4 + i;
        yr[c] = f2bf((vals[i] - mean) * rstd * w[c] + b[c]);
    }
}

// ---------------------------------------------------------------------------
// Repack qkv [row][3072] -> q[b,h,s,e] (scaled 1/8), k[b,h,t,e], vt[b,h,e,t]
// ---------------------------------------------------------------------------
__global__ __launch_bounds__(256) void repack_qkv(const u16* __restrict__ qkv,
                                                  u16* __restrict__ q,
                                                  u16* __restrict__ k,
                                                  u16* __restrict__ vt) {
    int i = blockIdx.x * 256 + threadIdx.x;      // over 4096*3072
    int row = i / 3072;
    int n   = i - row * 3072;
    int b = row >> 11, s = row & 2047;
    int sel = n >> 10;
    int hn  = n & 1023;
    int h = hn >> 6, e = hn & 63;
    int bh = b * NHEAD + h;
    u16 val = qkv[i];
    if (sel == 0) {
        q[((size_t)bh * SEQ + s) * HEAD + e] = f2bf(bf2f(val) * 0.125f);  // 1/sqrt(64)
    } else if (sel == 1) {
        k[((size_t)bh * SEQ + s) * HEAD + e] = val;
    } else {
        vt[((size_t)bh * HEAD + e) * SEQ + s] = val;
    }
}

// ---------------------------------------------------------------------------
// Generic bf16 WMMA GEMM: C[M][N] = A[M][K] * Bt[N][K]^T  (+ epilogue)
//   EPI 0: store bf16 | 1: +resid f32 | 2: +bias,gelu bf16 | 3: +bias+resid f32
// 256 threads, tile 128x128x32, 8 waves (4 in M x 2 in N).
// TDM double-buffered LDS staging: wave 0 issues tile i+1 before waiting on
// tile i (TENSORcnt completes in-order -> s_wait_tensorcnt 2 is exact).
// ---------------------------------------------------------------------------
template <int EPI>
__global__ __launch_bounds__(256) void gemm_bf16(const u16* __restrict__ A,
                                                 const u16* __restrict__ Bt,
                                                 float* __restrict__ outF,
                                                 u16* __restrict__ outB,
                                                 const float* __restrict__ resid,
                                                 const float* __restrict__ bias,
                                                 int M, int N, int K) {
    __shared__ u16 As[2][128 * 32];
    __shared__ u16 Bs[2][128 * 32];
    const int tid = threadIdx.x;
    const int w   = tid >> 5;
    const int l   = tid & 31;
    const int ln  = l & 15;
    const int kh  = l >> 4;
    const int wm  = w >> 1;          // 0..3 -> 32-row slab
    const int wn  = w & 1;           // 0..1 -> 64-col slab
    const int m0  = blockIdx.y * 128;
    const int n0  = blockIdx.x * 128;

    f32x8 c[2][4] = {};
    const int iters = K >> 5;

#if HAVE_TDM
    if (tid < 32) {
        tdm_load_2d(lds_off(&As[0][0]), A  + (size_t)m0 * K, 32, 128, (unsigned long long)K);
        tdm_load_2d(lds_off(&Bs[0][0]), Bt + (size_t)n0 * K, 32, 128, (unsigned long long)K);
    }
#endif

    for (int i = 0; i < iters; ++i) {
        const u16* Ab = As[i & 1];
        const u16* Bb = Bs[i & 1];
#if HAVE_TDM
        if (tid < 32) {
            if (i + 1 < iters) {
                tdm_load_2d(lds_off(&As[(i + 1) & 1][0]),
                            A + (size_t)m0 * K + (i + 1) * 32, 32, 128, (unsigned long long)K);
                tdm_load_2d(lds_off(&Bs[(i + 1) & 1][0]),
                            Bt + (size_t)n0 * K + (i + 1) * 32, 32, 128, (unsigned long long)K);
                __builtin_amdgcn_s_wait_tensorcnt(2);  // tile i done, i+1 in flight
            } else {
                __builtin_amdgcn_s_wait_tensorcnt(0);
            }
        }
        __syncthreads();
#else
        {   // fallback: vector staging
            const int lr = tid >> 1;
            const int lc = (tid & 1) * 16;
            __syncthreads();
            const uint4* sa = (const uint4*)(A + (size_t)(m0 + lr) * K + i * 32 + lc);
            uint4*       da = (uint4*)((u16*)Ab + lr * 32 + lc);
            da[0] = sa[0]; da[1] = sa[1];
            const uint4* sb = (const uint4*)(Bt + (size_t)(n0 + lr) * K + i * 32 + lc);
            uint4*       db = (uint4*)((u16*)Bb + lr * 32 + lc);
            db[0] = sb[0]; db[1] = sb[1];
            __syncthreads();
        }
#endif
        Frag a[2], b[4];
        for (int mt = 0; mt < 2; ++mt) {
            const u16* p = Ab + (wm * 32 + mt * 16 + ln) * 32;
            a[mt].q[0] = *(const uint4*)(p + 8 * kh);        // K = 8kh..+7
            a[mt].q[1] = *(const uint4*)(p + 16 + 8 * kh);   // K = 16+8kh..+7
        }
        for (int nt = 0; nt < 4; ++nt) {
            const u16* p = Bb + (wn * 64 + nt * 16 + ln) * 32;
            b[nt].q[0] = *(const uint4*)(p + 8 * kh);        // K = 8kh..+7
            b[nt].q[1] = *(const uint4*)(p + 16 + 8 * kh);   // K = 16+8kh..+7
        }
        for (int mt = 0; mt < 2; ++mt)
            for (int nt = 0; nt < 4; ++nt)
                c[mt][nt] = __builtin_amdgcn_wmma_f32_16x16x32_bf16(
                    false, a[mt].v, false, b[nt].v, (short)0, c[mt][nt], false, false);
        __syncthreads();
    }

    for (int mt = 0; mt < 2; ++mt)
        for (int nt = 0; nt < 4; ++nt)
            for (int r = 0; r < 8; ++r) {
                int row = m0 + wm * 32 + mt * 16 + r + 8 * kh;
                int col = n0 + wn * 64 + nt * 16 + ln;
                size_t idx = (size_t)row * N + col;
                float v = c[mt][nt][r];
                if (EPI == 0) {
                    outB[idx] = f2bf(v);
                } else if (EPI == 1) {
                    outF[idx] = v + resid[idx];
                } else if (EPI == 2) {
                    outB[idx] = f2bf(gelu_tanh(v + bias[col]));
                } else {
                    outF[idx] = v + bias[col] + resid[idx];
                }
            }
}

// ---------------------------------------------------------------------------
// Flash attention, causal. Grid (S/64, B*H), 128 threads = 4 waves.
// K/V tiles staged by TDM (uniform per-workgroup addresses).
// ---------------------------------------------------------------------------
__global__ __launch_bounds__(128) void flash_attn(const u16* __restrict__ Q,
                                                  const u16* __restrict__ Kg,
                                                  const u16* __restrict__ Vt,
                                                  u16* __restrict__ O) {
    const int qb  = blockIdx.x;   // query tile (64 rows)
    const int bh  = blockIdx.y;   // b*16+h
    const int b   = bh >> 4, hh = bh & 15;
    const int tid = threadIdx.x;
    const int w   = tid >> 5;     // wave 0..3 -> 16 query rows each
    const int l   = tid & 31;
    const int ln  = l & 15;
    const int kh  = l >> 4;

    __shared__ u16 Ks[64 * 64];       // [t][e]
    __shared__ u16 Vs[64 * 64];       // [e][t]
    __shared__ u16 Ps[4][16 * 64];    // per-wave P tile [m][t]

    // Persistent Q A-fragments (K dim = HEAD = 64 -> 2 wmma k-steps)
    const size_t qrow = ((size_t)bh * SEQ + qb * 64 + w * 16 + ln) * HEAD;
    Frag aq[2];
    for (int ks = 0; ks < 2; ++ks) {
        aq[ks].q[0] = *(const uint4*)(Q + qrow + ks * 32 + 8 * kh);
        aq[ks].q[1] = *(const uint4*)(Q + qrow + ks * 32 + 16 + 8 * kh);
    }

    f32x8 o[4] = {};
    float mrow[8], lrow[8];
    for (int r = 0; r < 8; ++r) { mrow[r] = -__builtin_inff(); lrow[r] = 0.0f; }

    for (int kt = 0; kt <= qb; ++kt) {
        __syncthreads();   // all waves done reading Ks/Vs from previous tile
#if HAVE_TDM
        if (tid < 32) {
            tdm_load_2d(lds_off(&Ks[0]), Kg + ((size_t)bh * SEQ + kt * 64) * HEAD,
                        64, 64, HEAD);
            tdm_load_2d(lds_off(&Vs[0]), Vt + (size_t)bh * HEAD * SEQ + kt * 64,
                        64, 64, SEQ);
            __builtin_amdgcn_s_wait_tensorcnt(0);
        }
#else
        {
            const int trow = tid >> 1;
            const int tcol = (tid & 1) * 32;
            const uint4* sk = (const uint4*)(Kg + ((size_t)bh * SEQ + kt * 64 + trow) * HEAD + tcol);
            uint4*       dk = (uint4*)(Ks + trow * 64 + tcol);
            dk[0] = sk[0]; dk[1] = sk[1]; dk[2] = sk[2]; dk[3] = sk[3];
            const uint4* sv = (const uint4*)(Vt + ((size_t)bh * HEAD + trow) * SEQ + kt * 64 + tcol);
            uint4*       dv = (uint4*)(Vs + trow * 64 + tcol);
            dv[0] = sv[0]; dv[1] = sv[1]; dv[2] = sv[2]; dv[3] = sv[3];
        }
#endif
        __syncthreads();

        // scores S = Q K^T (pre-scaled via Q)
        f32x8 s[4] = {};
        for (int ks = 0; ks < 2; ++ks)
            for (int nt = 0; nt < 4; ++nt) {
                Frag bk;
                const u16* p = Ks + (nt * 16 + ln) * 64 + ks * 32;
                bk.q[0] = *(const uint4*)(p + 8 * kh);
                bk.q[1] = *(const uint4*)(p + 16 + 8 * kh);
                s[nt] = __builtin_amdgcn_wmma_f32_16x16x32_bf16(
                    false, aq[ks].v, false, bk.v, (short)0, s[nt], false, false);
            }

        // causal mask on diagonal tile
        if (kt == qb) {
            for (int nt = 0; nt < 4; ++nt) {
                int col = kt * 64 + nt * 16 + ln;
                for (int r = 0; r < 8; ++r) {
                    int row = qb * 64 + w * 16 + r + 8 * kh;
                    if (col > row) s[nt][r] = -__builtin_inff();
                }
            }
        }

        // online softmax (rows live in the 16-lane group sharing bit4)
        float mnew[8], resc[8];
        for (int r = 0; r < 8; ++r) {
            float mx = fmaxf(fmaxf(s[0][r], s[1][r]), fmaxf(s[2][r], s[3][r]));
            mx = fmaxf(mx, __shfl_xor(mx, 1, 32));
            mx = fmaxf(mx, __shfl_xor(mx, 2, 32));
            mx = fmaxf(mx, __shfl_xor(mx, 4, 32));
            mx = fmaxf(mx, __shfl_xor(mx, 8, 32));
            mnew[r] = fmaxf(mrow[r], mx);
            resc[r] = __expf(mrow[r] - mnew[r]);
        }
        float rsum[8];
        for (int r = 0; r < 8; ++r) rsum[r] = 0.0f;
        for (int nt = 0; nt < 4; ++nt)
            for (int r = 0; r < 8; ++r) {
                float p = __expf(s[nt][r] - mnew[r]);
                s[nt][r] = p;
                rsum[r] += p;
            }
        for (int r = 0; r < 8; ++r) {
            float t = rsum[r];
            t += __shfl_xor(t, 1, 32);
            t += __shfl_xor(t, 2, 32);
            t += __shfl_xor(t, 4, 32);
            t += __shfl_xor(t, 8, 32);
            lrow[r] = lrow[r] * resc[r] + t;
            mrow[r] = mnew[r];
        }
        for (int nt = 0; nt < 4; ++nt)
            for (int r = 0; r < 8; ++r) o[nt][r] *= resc[r];

        // P (C-frag layout) -> per-wave LDS -> A-frag layout
        for (int nt = 0; nt < 4; ++nt)
            for (int r = 0; r < 8; ++r)
                Ps[w][(r + 8 * kh) * 64 + nt * 16 + ln] = f2bf(s[nt][r]);

        // O += P V   (K dim = 64 keys -> 2 wmma k-steps)
        for (int ks = 0; ks < 2; ++ks) {
            Frag ap;
            const u16* pp = Ps[w] + ln * 64 + ks * 32;
            ap.q[0] = *(const uint4*)(pp + 8 * kh);
            ap.q[1] = *(const uint4*)(pp + 16 + 8 * kh);
            for (int nt = 0; nt < 4; ++nt) {
                Frag bv;
                const u16* vp = Vs + (nt * 16 + ln) * 64 + ks * 32;
                bv.q[0] = *(const uint4*)(vp + 8 * kh);
                bv.q[1] = *(const uint4*)(vp + 16 + 8 * kh);
                o[nt] = __builtin_amdgcn_wmma_f32_16x16x32_bf16(
                    false, ap.v, false, bv.v, (short)0, o[nt], false, false);
            }
        }
    }

    // normalize and store: attnout[b*S + s][hh*64 + e] (bf16)
    for (int r = 0; r < 8; ++r) {
        float inv = 1.0f / lrow[r];
        int srow = qb * 64 + w * 16 + r + 8 * kh;
        size_t base = ((size_t)b * SEQ + srow) * D_MODEL + hh * HEAD;
        for (int nt = 0; nt < 4; ++nt)
            O[base + nt * 16 + ln] = f2bf(o[nt][r] * inv);
    }
}

// ---------------------------------------------------------------------------
// Host orchestration
// ---------------------------------------------------------------------------
extern "C" void kernel_launch(void* const* d_in, const int* in_sizes, int n_in,
                              void* d_out, int out_size, void* d_ws, size_t ws_size,
                              hipStream_t stream) {
    (void)in_sizes; (void)n_in; (void)out_size; (void)ws_size;
    const float* x    = (const float*)d_in[0];
    const float* ln1w = (const float*)d_in[1];
    const float* ln1b = (const float*)d_in[2];
    const float* wq   = (const float*)d_in[3];
    const float* wk   = (const float*)d_in[4];
    const float* wv   = (const float*)d_in[5];
    const float* wo   = (const float*)d_in[6];
    const float* ln2w = (const float*)d_in[7];
    const float* ln2b = (const float*)d_in[8];
    const float* w1   = (const float*)d_in[9];
    const float* b1   = (const float*)d_in[10];
    const float* w2   = (const float*)d_in[11];
    const float* b2   = (const float*)d_in[12];
    float* out = (float*)d_out;

    char* ws = (char*)d_ws;
    size_t off = 0;
    auto alloc = [&](size_t bytes) -> char* {
        char* p = ws + off;
        off += (bytes + 255) & ~(size_t)255;
        return p;
    };
    u16* wqkvT   = (u16*)alloc((size_t)3 * D_MODEL * D_MODEL * 2);
    u16* woT     = (u16*)alloc((size_t)D_MODEL * D_MODEL * 2);
    u16* w1b     = (u16*)alloc((size_t)D_FF * D_MODEL * 2);
    u16* w2b     = (u16*)alloc((size_t)D_MODEL * D_FF * 2);
    u16* ybuf    = (u16*)alloc((size_t)ROWS * D_MODEL * 2);
    u16* qkvbuf  = (u16*)alloc((size_t)ROWS * 3 * D_MODEL * 2);
    u16* qbuf    = (u16*)alloc((size_t)ROWS * D_MODEL * 2);
    u16* kbuf    = (u16*)alloc((size_t)ROWS * D_MODEL * 2);
    u16* vtbuf   = (u16*)alloc((size_t)ROWS * D_MODEL * 2);
    u16* attnout = (u16*)alloc((size_t)ROWS * D_MODEL * 2);
    float* x1    = (float*)alloc((size_t)ROWS * D_MODEL * 4);
    u16* y2buf   = (u16*)alloc((size_t)ROWS * D_MODEL * 2);
    u16* hbuf    = (u16*)alloc((size_t)ROWS * D_FF * 2);

    // weight prep
    prep_qkv_w<<<(3 * D_MODEL * D_MODEL) / 256, 256, 0, stream>>>(wq, wk, wv, wqkvT);
    prep_wo<<<(D_MODEL * D_MODEL) / 256, 256, 0, stream>>>(wo, woT);
    convert_f32_bf16<<<(D_FF * D_MODEL) / 256, 256, 0, stream>>>(w1, w1b, D_FF * D_MODEL);
    convert_f32_bf16<<<(D_MODEL * D_FF) / 256, 256, 0, stream>>>(w2, w2b, D_MODEL * D_FF);

    // attention sublayer
    ln_kernel<<<ROWS, 256, 0, stream>>>(x, ln1w, ln1b, ybuf);
    gemm_bf16<0><<<dim3((3 * D_MODEL) / 128, ROWS / 128), 256, 0, stream>>>(
        ybuf, wqkvT, nullptr, qkvbuf, nullptr, nullptr, ROWS, 3 * D_MODEL, D_MODEL);
    repack_qkv<<<(ROWS * 3 * D_MODEL) / 256, 256, 0, stream>>>(qkvbuf, qbuf, kbuf, vtbuf);
    flash_attn<<<dim3(SEQ / 64, BATCH * NHEAD), 128, 0, stream>>>(qbuf, kbuf, vtbuf, attnout);
    gemm_bf16<1><<<dim3(D_MODEL / 128, ROWS / 128), 256, 0, stream>>>(
        attnout, woT, x1, nullptr, x, nullptr, ROWS, D_MODEL, D_MODEL);

    // ffn sublayer
    ln_kernel<<<ROWS, 256, 0, stream>>>(x1, ln2w, ln2b, y2buf);
    gemm_bf16<2><<<dim3(D_FF / 128, ROWS / 128), 256, 0, stream>>>(
        y2buf, w1b, nullptr, hbuf, nullptr, b1, ROWS, D_FF, D_MODEL);
    gemm_bf16<3><<<dim3(D_MODEL / 128, ROWS / 128), 256, 0, stream>>>(
        hbuf, w2b, out, nullptr, x1, b2, ROWS, D_MODEL, D_FF);
}
